// OFAEncoderLayer_12945031430283
// MI455X (gfx1250) — matrix operand order repaired and checked by reading, not compiled, not run on hardware
//
#include <hip/hip_runtime.h>
#include <hip/hip_bf16.h>
#include <math.h>

typedef __attribute__((ext_vector_type(16))) __bf16 bf16x16;
typedef __attribute__((ext_vector_type(8)))  float  f32x8;

#define B_  4
#define L_  1024
#define C_  1024
#define H_  16
#define D_  64
#define FF_ 4096
#define MTOK (B_*L_)                 // 4096 tokens
__device__ __constant__ float SCALE_ = 0.08838834764831845f; // (2*64)^-0.5

// ---------------- WMMA helpers ----------------
__device__ inline f32x8 zero8() {
  f32x8 z;
#pragma unroll
  for (int i = 0; i < 8; i++) z[i] = 0.0f;
  return z;
}

// A-matrix 16x32 bf16 fragment: lane holds row (lane&15);
// lanes 0-15: K = k0+0..7 and k0+16..23 ; lanes 16-31: K = k0+8..15 and k0+24..31
__device__ inline bf16x16 frag_a(const __bf16* rowp, int k0, int hi) {
  const __bf16* p = rowp + k0 + hi * 8;
  bf16x16 f;
#pragma unroll
  for (int i = 0; i < 8; i++) { f[i] = p[i]; f[i + 8] = p[i + 16]; }
  return f;
}

// B-matrix 32x16 bf16 fragment: lane holds column (lane&15);
// lanes 0-15: K = k0+0..7 and k0+16..23 ; lanes 16-31: K = k0+8..15 and k0+24..31
__device__ inline bf16x16 frag_b(const __bf16* colp, int k0, int hi) {
  const __bf16* p = colp + k0 + hi * 8;
  bf16x16 f;
#pragma unroll
  for (int i = 0; i < 8; i++) { f[i] = p[i]; f[i + 8] = p[i + 16]; }
  return f;
}

__device__ inline f32x8 wmma_bf16(bf16x16 a, bf16x16 b, f32x8 c) {
  return __builtin_amdgcn_wmma_f32_16x16x32_bf16(false, a, false, b, (short)0, c,
                                                 false, false);
}

// ---------------- elementwise / LN kernels ----------------
__global__ void __launch_bounds__(256) cvt_f32_bf16(const float* __restrict__ in,
                                                    __bf16* __restrict__ out, int n) {
  int i = blockIdx.x * 256 + threadIdx.x;
  if (i < n) out[i] = (__bf16)in[i];
}

__device__ inline float blk_reduce256(float v, float* sb) {
  int t = threadIdx.x;
  sb[t] = v;
  __syncthreads();
  for (int s = 128; s > 0; s >>= 1) {
    if (t < s) sb[t] += sb[t + s];
    __syncthreads();
  }
  float r = sb[0];
  __syncthreads();
  return r;
}

// LayerNorm(fp32 row) -> bf16 row.  cols must be 1024 (4 per thread).
__global__ void __launch_bounds__(256) ln_f32_to_bf16(const float* __restrict__ in,
                                                      const float* __restrict__ g,
                                                      const float* __restrict__ bt,
                                                      __bf16* __restrict__ out) {
  __shared__ float sb[256];
  const int cols = C_;
  int row = blockIdx.x, t = threadIdx.x;
  const float* p = in + (size_t)row * cols + t * 4;
  float x[4], s = 0.f, q = 0.f;
#pragma unroll
  for (int i = 0; i < 4; i++) { x[i] = p[i]; s += x[i]; q += x[i] * x[i]; }
  float sum = blk_reduce256(s, sb);
  float sq  = blk_reduce256(q, sb);
  float mu  = sum / cols;
  float var = sq / cols - mu * mu;
  float rs  = rsqrtf(var + 1e-5f);
  __bf16* o = out + (size_t)row * cols + t * 4;
#pragma unroll
  for (int i = 0; i < 4; i++)
    o[i] = (__bf16)((x[i] - mu) * rs * g[t * 4 + i] + bt[t * 4 + i]);
}

// x1 = residual + LayerNorm(proj)   (all fp32, cols = 1024)
__global__ void __launch_bounds__(256) midln_res(const float* __restrict__ in,
                                                 const float* __restrict__ resid,
                                                 const float* __restrict__ g,
                                                 const float* __restrict__ bt,
                                                 float* __restrict__ out) {
  __shared__ float sb[256];
  const int cols = C_;
  int row = blockIdx.x, t = threadIdx.x;
  const float* p = in + (size_t)row * cols + t * 4;
  const float* r = resid + (size_t)row * cols + t * 4;
  float x[4], s = 0.f, q = 0.f;
#pragma unroll
  for (int i = 0; i < 4; i++) { x[i] = p[i]; s += x[i]; q += x[i] * x[i]; }
  float sum = blk_reduce256(s, sb);
  float sq  = blk_reduce256(q, sb);
  float mu  = sum / cols;
  float var = sq / cols - mu * mu;
  float rst = rsqrtf(var + 1e-5f);
  float* o = out + (size_t)row * cols + t * 4;
#pragma unroll
  for (int i = 0; i < 4; i++)
    o[i] = r[i] + (x[i] - mu) * rst * g[t * 4 + i] + bt[t * 4 + i];
}

// in-place LayerNorm over FF (=4096) on a bf16 buffer (16 per thread)
__global__ void __launch_bounds__(256) ln_bf16_inplace(__bf16* __restrict__ buf,
                                                       const float* __restrict__ g,
                                                       const float* __restrict__ bt) {
  __shared__ float sb[256];
  const int cols = FF_;
  int row = blockIdx.x, t = threadIdx.x;
  __bf16* p = buf + (size_t)row * cols + t * 16;
  float x[16], s = 0.f, q = 0.f;
#pragma unroll
  for (int i = 0; i < 16; i++) { x[i] = (float)p[i]; s += x[i]; q += x[i] * x[i]; }
  float sum = blk_reduce256(s, sb);
  float sq  = blk_reduce256(q, sb);
  float mu  = sum / cols;
  float var = sq / cols - mu * mu;
  float rst = rsqrtf(var + 1e-5f);
#pragma unroll
  for (int i = 0; i < 16; i++)
    p[i] = (__bf16)((x[i] - mu) * rst * g[t * 16 + i] + bt[t * 16 + i]);
}

// ---------------- WMMA GEMM: D[M,N] = A[M,K] @ W[N,K]^T + bias ----------------
// Software-pipelined: next k-step fragments are loaded before issuing the
// current step's 4 WMMAs so global-load latency overlaps tensor execution.
// mode 0: outF[m*N+n] = v
// mode 1: outB head layout (B,H,L,D)      (Q, K projections)
// mode 2: outB transposed head (B,H,D,L)  (V projection)
// mode 3: outB[m*N+n] = bf16(gelu(v))     (FFN up)
// mode 4: outF[m*N+n] = v + res[m*N+n]    (FFN down -> d_out)
__global__ void __launch_bounds__(256) gemm_wmma(const __bf16* __restrict__ A,
                                                 const __bf16* __restrict__ W,
                                                 const float* __restrict__ bias,
                                                 float* __restrict__ outF,
                                                 __bf16* __restrict__ outB,
                                                 const float* __restrict__ res,
                                                 int M, int N, int K, int mode) {
  int lane = threadIdx.x & 31;
  int wid  = blockIdx.x * (blockDim.x >> 5) + (threadIdx.x >> 5);
  int tilesN = N >> 6;
  int tmi = wid / tilesN, tni = wid % tilesN;
  if (tmi * 16 >= M) return;
  int hi = lane >> 4, ln = lane & 15;

  const __bf16* arow = A + (size_t)(tmi * 16 + ln) * K;
  const __bf16* w0 = W + (size_t)(tni * 64 +  0 + ln) * K;
  const __bf16* w1 = W + (size_t)(tni * 64 + 16 + ln) * K;
  const __bf16* w2 = W + (size_t)(tni * 64 + 32 + ln) * K;
  const __bf16* w3 = W + (size_t)(tni * 64 + 48 + ln) * K;

  f32x8 c0 = zero8(), c1 = zero8(), c2 = zero8(), c3 = zero8();

  // prologue: fragments for k0 = 0
  bf16x16 a  = frag_a(arow, 0, hi);
  bf16x16 b0 = frag_b(w0, 0, hi);
  bf16x16 b1 = frag_b(w1, 0, hi);
  bf16x16 b2 = frag_b(w2, 0, hi);
  bf16x16 b3 = frag_b(w3, 0, hi);

  for (int k0 = 32; k0 < K; k0 += 32) {
    __builtin_prefetch(arow + k0 + 256, 0, 3);
    // issue next-step loads first (overlap with WMMAs below)
    bf16x16 an  = frag_a(arow, k0, hi);
    bf16x16 nb0 = frag_b(w0, k0, hi);
    bf16x16 nb1 = frag_b(w1, k0, hi);
    bf16x16 nb2 = frag_b(w2, k0, hi);
    bf16x16 nb3 = frag_b(w3, k0, hi);
    c0 = wmma_bf16(a, b0, c0);
    c1 = wmma_bf16(a, b1, c1);
    c2 = wmma_bf16(a, b2, c2);
    c3 = wmma_bf16(a, b3, c3);
    a = an; b0 = nb0; b1 = nb1; b2 = nb2; b3 = nb3;
  }
  // tail k-step
  c0 = wmma_bf16(a, b0, c0);
  c1 = wmma_bf16(a, b1, c1);
  c2 = wmma_bf16(a, b2, c2);
  c3 = wmma_bf16(a, b3, c3);

  f32x8 cc[4] = {c0, c1, c2, c3};
#pragma unroll
  for (int j = 0; j < 4; j++) {
    int n = tni * 64 + j * 16 + ln;
    float bn = bias[n];
#pragma unroll
    for (int i = 0; i < 8; i++) {
      int m = tmi * 16 + i + 8 * hi;
      float v = cc[j][i] + bn;
      if (mode == 0) {
        outF[(size_t)m * N + n] = v;
      } else if (mode == 1) {
        int b = m >> 10, l = m & (L_ - 1), h = n >> 6, d = n & (D_ - 1);
        outB[(((size_t)(b * H_ + h)) * L_ + l) * D_ + d] = (__bf16)v;
      } else if (mode == 2) {
        int b = m >> 10, l = m & (L_ - 1), h = n >> 6, d = n & (D_ - 1);
        outB[(((size_t)(b * H_ + h)) * D_ + d) * L_ + l] = (__bf16)v;
      } else if (mode == 3) {
        float gl = 0.5f * v * (1.0f + erff(v * 0.70710678118654752f));
        outB[(size_t)m * N + n] = (__bf16)gl;
      } else {
        outF[(size_t)m * N + n] = v + res[(size_t)m * N + n];
      }
    }
  }
}

// ---------------- flash attention: one wave per 16 query rows per (b,h) ------
__global__ void __launch_bounds__(32) attn_flash(const __bf16* __restrict__ q,
                                                 const __bf16* __restrict__ kk,
                                                 const __bf16* __restrict__ vt,
                                                 const float* __restrict__ bias,
                                                 const float* __restrict__ mask,
                                                 const float* __restrict__ c_attn,
                                                 __bf16* __restrict__ out) {
  __shared__ __bf16 ptile[16 * 32];
  int lane = threadIdx.x;
  int hi = lane >> 4, ln = lane & 15;
  int wid  = blockIdx.x;
  int rblk = wid & (L_ / 16 - 1);
  int bh   = wid >> 6;          // (L_/16)=64 blocks per (b,h)
  int b = bh >> 4, h = bh & (H_ - 1);
  int row = rblk * 16 + ln;

  const __bf16* qrow = q + ((size_t)bh * L_ + row) * D_;
  bf16x16 a0 = frag_a(qrow, 0, hi);
  bf16x16 a1 = frag_a(qrow, 32, hi);
  const __bf16* kbase = kk + (size_t)bh * L_ * D_;
  const __bf16* vbase = vt + (size_t)bh * D_ * L_;
  const float*  bbase = bias + ((size_t)bh * L_ + rblk * 16) * L_;
  const float*  mbase = mask + ((size_t)b * L_ + rblk * 16) * L_;

  float mst[8], lst[8];
#pragma unroll
  for (int i = 0; i < 8; i++) { mst[i] = -INFINITY; lst[i] = 0.0f; }
  f32x8 o0 = zero8(), o1 = zero8(), o2 = zero8(), o3 = zero8();

  for (int sb = 0; sb < L_ / 32; sb++) {
    int s0 = sb * 32;
    const __bf16* kr0 = kbase + (size_t)(s0 + ln) * D_;
    const __bf16* kr1 = kbase + (size_t)(s0 + 16 + ln) * D_;
    f32x8 sA = zero8(), sB = zero8();
    sA = wmma_bf16(a0, frag_b(kr0, 0, hi), sA);
    sA = wmma_bf16(a1, frag_b(kr0, 32, hi), sA);
    sB = wmma_bf16(a0, frag_b(kr1, 0, hi), sB);
    sB = wmma_bf16(a1, frag_b(kr1, 32, hi), sB);

    float e0[8], e1[8];
#pragma unroll
    for (int i = 0; i < 8; i++) {
      int m = i + 8 * hi;                         // C-layout row for this half
      size_t boff = (size_t)m * L_ + s0 + ln;
      float x0 = sA[i] * SCALE_ + bbase[boff]      + mbase[boff];
      float x1 = sB[i] * SCALE_ + bbase[boff + 16] + mbase[boff + 16];
      float tmx = fmaxf(x0, x1);
#pragma unroll
      for (int w = 8; w >= 1; w >>= 1) tmx = fmaxf(tmx, __shfl_xor(tmx, w, 32));
      float mnew  = fmaxf(mst[i], tmx);
      float alpha = __expf(mst[i] - mnew);
      mst[i] = mnew;
      float p0 = __expf(x0 - mnew), p1 = __expf(x1 - mnew);
      float rs = p0 + p1;
#pragma unroll
      for (int w = 8; w >= 1; w >>= 1) rs += __shfl_xor(rs, w, 32);
      lst[i] = lst[i] * alpha + rs;
      o0[i] *= alpha; o1[i] *= alpha; o2[i] *= alpha; o3[i] *= alpha;
      e0[i] = p0; e1[i] = p1;
    }
    __syncthreads();
#pragma unroll
    for (int i = 0; i < 8; i++) {
      int m = i + 8 * hi;
      ptile[m * 32 + ln]      = (__bf16)e0[i];
      ptile[m * 32 + 16 + ln] = (__bf16)e1[i];
    }
    __syncthreads();
    bf16x16 pa = frag_a(&ptile[ln * 32], 0, hi);   // P in A-layout (16x32)
    o0 = wmma_bf16(pa, frag_b(vbase + (size_t)( 0 + ln) * L_ + s0, 0, hi), o0);
    o1 = wmma_bf16(pa, frag_b(vbase + (size_t)(16 + ln) * L_ + s0, 0, hi), o1);
    o2 = wmma_bf16(pa, frag_b(vbase + (size_t)(32 + ln) * L_ + s0, 0, hi), o2);
    o3 = wmma_bf16(pa, frag_b(vbase + (size_t)(48 + ln) * L_ + s0, 0, hi), o3);
  }

  float ca = c_attn[h];
  f32x8 oo[4] = {o0, o1, o2, o3};
#pragma unroll
  for (int j = 0; j < 4; j++)
#pragma unroll
    for (int i = 0; i < 8; i++) {
      int m = i + 8 * hi;
      float v = oo[j][i] / lst[i] * ca;
      out[((size_t)b * L_ + rblk * 16 + m) * C_ + h * D_ + j * 16 + ln] = (__bf16)v;
    }
}

// ---------------- host orchestration ----------------
extern "C" void kernel_launch(void* const* d_in, const int* in_sizes, int n_in,
                              void* d_out, int out_size, void* d_ws, size_t ws_size,
                              hipStream_t stream) {
  const float* x        = (const float*)d_in[0];
  const float* abias    = (const float*)d_in[1];
  const float* amask    = (const float*)d_in[2];
  const float* Wq = (const float*)d_in[3];  const float* bq = (const float*)d_in[4];
  const float* Wk = (const float*)d_in[5];  const float* bk = (const float*)d_in[6];
  const float* Wv = (const float*)d_in[7];  const float* bv = (const float*)d_in[8];
  const float* Wo = (const float*)d_in[9];  const float* bo = (const float*)d_in[10];
  const float* c_attn = (const float*)d_in[11];
  const float* W1 = (const float*)d_in[12]; const float* b1 = (const float*)d_in[13];
  const float* W2 = (const float*)d_in[14]; const float* b2 = (const float*)d_in[15];
  const float* aln_g  = (const float*)d_in[16]; const float* aln_b  = (const float*)d_in[17];
  const float* amid_g = (const float*)d_in[18]; const float* amid_b = (const float*)d_in[19];
  const float* fln_g  = (const float*)d_in[20]; const float* fln_b  = (const float*)d_in[21];
  const float* fmid_g = (const float*)d_in[22]; const float* fmid_b = (const float*)d_in[23];
  float* out = (float*)d_out;

  const size_t MB = (size_t)1 << 20;
  char* ws = (char*)d_ws;
  __bf16* wq_b = (__bf16*)(ws + 0 * MB);     // 2 MB each (C*C bf16)
  __bf16* wk_b = (__bf16*)(ws + 2 * MB);
  __bf16* wv_b = (__bf16*)(ws + 4 * MB);
  __bf16* wo_b = (__bf16*)(ws + 6 * MB);
  __bf16* w1_b = (__bf16*)(ws + 8 * MB);     // 8 MB (FF*C)
  __bf16* w2_b = (__bf16*)(ws + 16 * MB);    // 8 MB (C*FF)
  float*  x1   = (float*)(ws + 24 * MB);     // 16 MB (MTOK*C f32)
  __bf16* hbuf = (__bf16*)(ws + 40 * MB);    // 8 MB  (MTOK*C bf16)
  __bf16* qb   = (__bf16*)(ws + 48 * MB);    // 8 MB
  __bf16* kb   = (__bf16*)(ws + 56 * MB);    // 8 MB
  __bf16* vtb  = (__bf16*)(ws + 64 * MB);    // 8 MB
  __bf16* attb = (__bf16*)(ws + 72 * MB);    // 8 MB
  float*  proj = (float*)(ws + 80 * MB);     // 16 MB
  __bf16* ff1  = (__bf16*)(ws + 48 * MB);    // 32 MB, reuses q/k/vt after attn

  // 1) weights -> bf16
  cvt_f32_bf16<<<(C_ * C_) / 256, 256, 0, stream>>>(Wq, wq_b, C_ * C_);
  cvt_f32_bf16<<<(C_ * C_) / 256, 256, 0, stream>>>(Wk, wk_b, C_ * C_);
  cvt_f32_bf16<<<(C_ * C_) / 256, 256, 0, stream>>>(Wv, wv_b, C_ * C_);
  cvt_f32_bf16<<<(C_ * C_) / 256, 256, 0, stream>>>(Wo, wo_b, C_ * C_);
  cvt_f32_bf16<<<(FF_ * C_) / 256, 256, 0, stream>>>(W1, w1_b, FF_ * C_);
  cvt_f32_bf16<<<(FF_ * C_) / 256, 256, 0, stream>>>(W2, w2_b, FF_ * C_);

  // 2) pre-attn LN
  ln_f32_to_bf16<<<MTOK, 256, 0, stream>>>(x, aln_g, aln_b, hbuf);

  // 3) QKV projections (WMMA): tiles = (MTOK/16)*(C/64) = 4096 waves -> 512 blocks
  gemm_wmma<<<512, 256, 0, stream>>>(hbuf, wq_b, bq, nullptr, qb, nullptr,
                                     MTOK, C_, C_, 1);
  gemm_wmma<<<512, 256, 0, stream>>>(hbuf, wk_b, bk, nullptr, kb, nullptr,
                                     MTOK, C_, C_, 1);
  gemm_wmma<<<512, 256, 0, stream>>>(hbuf, wv_b, bv, nullptr, vtb, nullptr,
                                     MTOK, C_, C_, 2);

  // 4) flash attention: B*H*(L/16) = 4096 waves
  attn_flash<<<B_ * H_ * (L_ / 16), 32, 0, stream>>>(qb, kb, vtb, abias, amask,
                                                     c_attn, attb);

  // 5) output projection
  gemm_wmma<<<512, 256, 0, stream>>>(attb, wo_b, bo, proj, nullptr, nullptr,
                                     MTOK, C_, C_, 0);

  // 6) mid-LN + residual
  midln_res<<<MTOK, 256, 0, stream>>>(proj, x, amid_g, amid_b, x1);

  // 7) FFN pre-LN
  ln_f32_to_bf16<<<MTOK, 256, 0, stream>>>(x1, fln_g, fln_b, hbuf);

  // 8) FFN up + GELU: tiles = 256*64 = 16384 waves -> 2048 blocks
  gemm_wmma<<<2048, 256, 0, stream>>>(hbuf, w1_b, b1, nullptr, ff1, nullptr,
                                      MTOK, FF_, C_, 3);

  // 9) FFN mid-LN (in place on bf16)
  ln_bf16_inplace<<<MTOK, 256, 0, stream>>>(ff1, fmid_g, fmid_b);

  // 10) FFN down + bias + residual -> d_out
  gemm_wmma<<<512, 256, 0, stream>>>(ff1, w2_b, b2, out, nullptr, x1,
                                     MTOK, C_, FF_, 4);

  (void)in_sizes; (void)n_in; (void)out_size; (void)ws_size;
}
